// EAGRR_42674795053753
// MI455X (gfx1250) — compile-verified, matrix operand order checked by reading, not run on hardware
//
#include <hip/hip_runtime.h>
#include <hip/hip_bf16.h>

// ---------------- problem constants ----------------
#define B_     2
#define N_     512
#define FDIM_  1024
#define HEADS_ 4
#define DH_    256
#define AH_    256
#define HID_   512
#define OUT_   128
#define NCLS_  10
#define NN_    (N_ * N_)           // 262144
#define KEEP_  209715              // int(0.8 * N*N)
#define RANK_  (NN_ - KEEP_)       // 52429: 0-based ascending rank of threshold
#define INV_TEMP_ 2.0f             // 1/TEMP, TEMP=0.5

typedef __attribute__((ext_vector_type(16))) __bf16 v16bf;
typedef __attribute__((ext_vector_type(2)))  __bf16 bf16x2;
typedef __attribute__((ext_vector_type(8)))  float  v8f;

__device__ __forceinline__ unsigned pack_bf16x2(float lo, float hi)
{
    bf16x2 v;
    v.x = (__bf16)lo;
    v.y = (__bf16)hi;
    return __builtin_bit_cast(unsigned, v);   // single v_cvt_pk_bf16_f32
}

union FragCast { uint4 u[2]; v16bf v; };

// =====================================================================
// Generic batched GEMM:  C = A(MxK) * B + optional bias over N columns.
// TRANSB==0: B is (K x N) row-major, ldb          ("NN")
// TRANSB==1: B is (N x K) row-major, ldb          ("NT", A*B^T)
// 256 threads = 8 waves; 128x64 output tile; BK=32.
// LDS holds bf16 tiles as packed pairs, K-pair-major, 20-dword row
// stride: each fragment = two 16B-aligned ds_load_b128, zero inner-loop
// conversions. Global loads for tile kt+1 are software-pipelined behind
// the WMMAs of tile kt; tile kt+2 is prefetched (global_prefetch).
// Requires M%128==0, N%64==0, K%32==0 (true for all launches below).
// =====================================================================
template<int TRANSB, bool BIAS>
__global__ __launch_bounds__(256)
void gemm_wmma_bf16(const float* __restrict__ Ag, const float* __restrict__ Bg,
                    float* __restrict__ Cg, const float* __restrict__ bias,
                    int M, int Nn, int K, int lda, int ldb, int ldc,
                    int zInner,
                    long long sAo, long long sAi,
                    long long sBo, long long sBi,
                    long long sCo, long long sCi)
{
    (void)M; (void)Nn;
    const int z  = blockIdx.z;
    const int zo = z / zInner;
    const int zi = z - zo * zInner;
    const float* A  = Ag + zo * sAo + zi * sAi;
    const float* Bm = Bg + zo * sBo + zi * sBi;
    float*       C  = Cg + zo * sCo + zi * sCi;

    // packed bf16 pairs: [row][K/2], stride 20 dwords (16B-aligned uint4 views)
    __shared__ unsigned As2[128][20];
    __shared__ unsigned Bs2[64][20];

    const int tid  = threadIdx.x;
    const int wave = tid >> 5;
    const int lane = tid & 31;
    const int ml   = lane & 15;
    const int hi2  = (lane >> 4) & 1;

    const int row0 = blockIdx.y * 128;
    const int col0 = blockIdx.x * 64;

    v8f acc[4] = {};

    // ---- per-thread staging indices ----
    // A: 128x32 floats, 4 float4 per thread
    // B (NN): micro 2(k) x 4(c) per thread ; B (NT): 2 float4 per thread
    const int aR[4] = { ((0 * 256 + tid) * 4) >> 5, ((1 * 256 + tid) * 4) >> 5,
                        ((2 * 256 + tid) * 4) >> 5, ((3 * 256 + tid) * 4) >> 5 };
    const int aK[4] = { ((0 * 256 + tid) * 4) & 31, ((1 * 256 + tid) * 4) & 31,
                        ((2 * 256 + tid) * 4) & 31, ((3 * 256 + tid) * 4) & 31 };
    const int bkp = tid >> 4;          // NN: k-pair 0..15
    const int bc  = (tid & 15) << 2;   // NN: col group *4
    const int bC[2] = { ((0 * 256 + tid) * 4) >> 5, ((1 * 256 + tid) * 4) >> 5 }; // NT
    const int bK[2] = { ((0 * 256 + tid) * 4) & 31, ((1 * 256 + tid) * 4) & 31 };

    float4 ra[4], rb[2];

    // ---- prologue fetch: tile 0 ----
    {
        const int k0 = 0;
        #pragma unroll
        for (int i = 0; i < 4; ++i)
            ra[i] = *reinterpret_cast<const float4*>(
                &A[(size_t)(row0 + aR[i]) * lda + k0 + aK[i]]);
        if (TRANSB == 0) {
            rb[0] = *reinterpret_cast<const float4*>(
                &Bm[(size_t)(k0 + 2 * bkp) * ldb + col0 + bc]);
            rb[1] = *reinterpret_cast<const float4*>(
                &Bm[(size_t)(k0 + 2 * bkp + 1) * ldb + col0 + bc]);
        } else {
            #pragma unroll
            for (int i = 0; i < 2; ++i)
                rb[i] = *reinterpret_cast<const float4*>(
                    &Bm[(size_t)(col0 + bC[i]) * ldb + k0 + bK[i]]);
        }
    }

    const int kTiles = K >> 5;
    for (int kt = 0; kt < kTiles; ++kt) {
        // ---- pack current registers into bf16 LDS tiles ----
        #pragma unroll
        for (int i = 0; i < 4; ++i) {
            As2[aR[i]][(aK[i] >> 1) + 0] = pack_bf16x2(ra[i].x, ra[i].y);
            As2[aR[i]][(aK[i] >> 1) + 1] = pack_bf16x2(ra[i].z, ra[i].w);
        }
        if (TRANSB == 0) {
            #pragma unroll
            for (int q = 0; q < 4; ++q)
                Bs2[bc + q][bkp] = pack_bf16x2((&rb[0].x)[q], (&rb[1].x)[q]);
        } else {
            #pragma unroll
            for (int i = 0; i < 2; ++i) {
                Bs2[bC[i]][(bK[i] >> 1) + 0] = pack_bf16x2(rb[i].x, rb[i].y);
                Bs2[bC[i]][(bK[i] >> 1) + 1] = pack_bf16x2(rb[i].z, rb[i].w);
            }
        }
        __syncthreads();

        // ---- pipelined global fetch of tile kt+1 ----
        if (kt + 1 < kTiles) {
            const int k0 = (kt + 1) << 5;
            #pragma unroll
            for (int i = 0; i < 4; ++i)
                ra[i] = *reinterpret_cast<const float4*>(
                    &A[(size_t)(row0 + aR[i]) * lda + k0 + aK[i]]);
            if (TRANSB == 0) {
                rb[0] = *reinterpret_cast<const float4*>(
                    &Bm[(size_t)(k0 + 2 * bkp) * ldb + col0 + bc]);
                rb[1] = *reinterpret_cast<const float4*>(
                    &Bm[(size_t)(k0 + 2 * bkp + 1) * ldb + col0 + bc]);
            } else {
                #pragma unroll
                for (int i = 0; i < 2; ++i)
                    rb[i] = *reinterpret_cast<const float4*>(
                        &Bm[(size_t)(col0 + bC[i]) * ldb + k0 + bK[i]]);
            }
        }
        // ---- prefetch tile kt+2 (global_prefetch_b8) ----
        if (kt + 2 < kTiles) {
            const int kp0 = (kt + 2) << 5;
            __builtin_prefetch(&A[(size_t)(row0 + (tid & 127)) * lda + kp0], 0, 1);
            if (TRANSB == 0)
                __builtin_prefetch(&Bm[(size_t)(kp0 + (tid & 31)) * ldb + col0], 0, 1);
            else
                __builtin_prefetch(&Bm[(size_t)(col0 + (tid & 63)) * ldb + kp0], 0, 1);
        }

        // ---- fragments: two ds_load_b128 each, bit-cast to v16bf ----
        const int rowA = (wave << 4) + ml;
        FragCast fa;
        fa.u[0] = *reinterpret_cast<const uint4*>(&As2[rowA][4 * hi2]);
        fa.u[1] = *reinterpret_cast<const uint4*>(&As2[rowA][8 + 4 * hi2]);
        const v16bf af = fa.v;

        #pragma unroll
        for (int t = 0; t < 4; ++t) {
            const int rowB = (t << 4) + ml;
            FragCast fb;
            fb.u[0] = *reinterpret_cast<const uint4*>(&Bs2[rowB][4 * hi2]);
            fb.u[1] = *reinterpret_cast<const uint4*>(&Bs2[rowB][8 + 4 * hi2]);
            acc[t] = __builtin_amdgcn_wmma_f32_16x16x32_bf16(
                false, af, false, fb.v, (short)0, acc[t], false, false);
        }
        __syncthreads();
    }

    // ---- store C per documented 16x16 f32 D layout ----
    #pragma unroll
    for (int t = 0; t < 4; ++t) {
        int c = col0 + (t << 4) + ml;
        float bv = BIAS ? bias[c] : 0.0f;
        #pragma unroll
        for (int g = 0; g < 8; ++g) {
            int r = row0 + (wave << 4) + g + (hi2 << 3);
            C[(size_t)r * ldc + c] = acc[t][g] + bv;
        }
    }
}

// =====================================================================
// Attention scores: s_i[b,h,n] = hp[b,n,h*DH:] . a_i[h],  s_j likewise.
// =====================================================================
__global__ __launch_bounds__(256)
void attn_scores_k(const float* __restrict__ hp, const float* __restrict__ attn,
                   float* __restrict__ s_i, float* __restrict__ s_j)
{
    __shared__ float r1[256], r2[256];
    const int idx = blockIdx.x;                 // b*H*N + h*N + n
    const int n = idx % N_;
    const int h = (idx / N_) % HEADS_;
    const int b = idx / (N_ * HEADS_);
    const int d = threadIdx.x;

    float v  = hp[((size_t)b * N_ + n) * FDIM_ + h * DH_ + d];
    float ai = attn[h * (2 * DH_) + d];
    float aj = attn[h * (2 * DH_) + DH_ + d];
    r1[d] = v * ai; r2[d] = v * aj;
    __syncthreads();
    for (int s = 128; s > 0; s >>= 1) {
        if (d < s) { r1[d] += r1[d + s]; r2[d] += r2[d + s]; }
        __syncthreads();
    }
    if (d == 0) { s_i[idx] = r1[0]; s_j[idx] = r2[0]; }
}

// =====================================================================
// alpha[b,h,i,:] = softmax_j( leakyrelu(s_i + s_j) ). 512 threads/row.
// =====================================================================
__global__ __launch_bounds__(512)
void attn_softmax_k(const float* __restrict__ s_i, const float* __restrict__ s_j,
                    float* __restrict__ alpha)
{
    __shared__ float red[512];
    const int row  = blockIdx.x;            // (b*H+h)*N + i
    const int base = (row / N_) * N_;
    const int j = threadIdx.x;

    float e = s_i[row] + s_j[base + j];
    e = (e >= 0.0f) ? e : 0.2f * e;

    red[j] = e; __syncthreads();
    for (int s = 256; s > 0; s >>= 1) {
        if (j < s) red[j] = fmaxf(red[j], red[j + s]);
        __syncthreads();
    }
    float mx = red[0]; __syncthreads();
    float p = __expf(e - mx);
    red[j] = p; __syncthreads();
    for (int s = 256; s > 0; s >>= 1) {
        if (j < s) red[j] += red[j + s];
        __syncthreads();
    }
    alpha[(size_t)row * N_ + j] = p * (1.0f / red[0]);
}

// =====================================================================
// pjT[b,a,n] = pj[b,n,a]   (tiled 32x32 transpose)
// =====================================================================
__global__ __launch_bounds__(256)
void transpose_pj_k(const float* __restrict__ pj, float* __restrict__ pjT)
{
    __shared__ float tile[32][33];
    const int n0 = blockIdx.x * 32, a0 = blockIdx.y * 32, b = blockIdx.z;
    const int tx = threadIdx.x, ty = threadIdx.y;
    for (int r = ty; r < 32; r += 8)
        tile[r][tx] = pj[((size_t)b * N_ + n0 + r) * AH_ + a0 + tx];
    __syncthreads();
    for (int r = ty; r < 32; r += 8)
        pjT[((size_t)b * AH_ + a0 + r) * N_ + n0 + tx] = tile[tx][r];
}

// =====================================================================
// edge_scores[b,i,j] = b2 + sum_a relu(pi[b,i,a] + pjT[b,a,j] + b1[a])*w2[a]
// =====================================================================
__global__ __launch_bounds__(256)
void edge_scores_k(const float* __restrict__ pi, const float* __restrict__ pjT,
                   const float* __restrict__ b1, const float* __restrict__ w2,
                   const float* __restrict__ b2, float* __restrict__ es)
{
    __shared__ float pis[AH_], b1s[AH_], w2s[AH_];
    const int i = blockIdx.x, b = blockIdx.y, t = threadIdx.x;
    pis[t] = pi[((size_t)b * N_ + i) * AH_ + t];
    b1s[t] = b1[t];
    w2s[t] = w2[t];
    __syncthreads();
    const float bb = b2[0];
    const float* base = pjT + (size_t)b * AH_ * N_;
    for (int j = t; j < N_; j += 256) {
        float acc = 0.0f;
        #pragma unroll 8
        for (int a = 0; a < AH_; ++a) {
            float v = pis[a] + base[(size_t)a * N_ + j] + b1s[a];
            acc = fmaf(fmaxf(v, 0.0f), w2s[a], acc);
        }
        es[((size_t)b * N_ + i) * N_ + j] = acc + bb;
    }
}

// =====================================================================
// soft_mask[b,:] = softmax(es[b,:] / TEMP) over N*N.
// =====================================================================
__global__ __launch_bounds__(1024)
void flat_softmax_k(const float* __restrict__ es, float* __restrict__ soft)
{
    __shared__ float red[1024];
    const int b = blockIdx.x, t = threadIdx.x;
    const float* x = es + (size_t)b * NN_;
    float* y = soft + (size_t)b * NN_;

    float mx = -3.4e38f;
    for (int i = t; i < NN_; i += 1024) mx = fmaxf(mx, x[i] * INV_TEMP_);
    red[t] = mx; __syncthreads();
    for (int s = 512; s > 0; s >>= 1) {
        if (t < s) red[t] = fmaxf(red[t], red[t + s]);
        __syncthreads();
    }
    mx = red[0]; __syncthreads();

    float sm = 0.0f;
    for (int i = t; i < NN_; i += 1024) sm += __expf(x[i] * INV_TEMP_ - mx);
    red[t] = sm; __syncthreads();
    for (int s = 512; s > 0; s >>= 1) {
        if (t < s) red[t] += red[t + s];
        __syncthreads();
    }
    const float inv = 1.0f / red[0];

    for (int i = t; i < NN_; i += 1024)
        y[i] = __expf(x[i] * INV_TEMP_ - mx) * inv;
}

// =====================================================================
// Exact rank selection via two-pass 65536-bin histogram in LDS (256KB
// dynamic LDS, legal on CDNA5's 320KB/WGP).
// =====================================================================
__global__ __launch_bounds__(1024)
void select_thr_k(const float* __restrict__ soft, float* __restrict__ thr)
{
    extern __shared__ unsigned hist[];   // 65536 bins
    __shared__ unsigned psum[1024];
    __shared__ unsigned sel_hi, rem_s;
    const int b = blockIdx.x, t = threadIdx.x;
    const float* x = soft + (size_t)b * NN_;

    for (int i = t; i < 65536; i += 1024) hist[i] = 0u;
    __syncthreads();
    for (int i = t; i < NN_; i += 1024) {
        unsigned bits = __float_as_uint(x[i]);
        atomicAdd(&hist[bits >> 16], 1u);
    }
    __syncthreads();
    unsigned s = 0;
    for (int i = 0; i < 64; ++i) s += hist[t * 64 + i];
    psum[t] = s; __syncthreads();
    if (t == 0) {
        unsigned cum = 0; int chunk = 0;
        for (; chunk < 1024; ++chunk) {
            if (cum + psum[chunk] > (unsigned)RANK_) break;
            cum += psum[chunk];
        }
        int bin = chunk * 64;
        for (;; ++bin) {
            if (cum + hist[bin] > (unsigned)RANK_) break;
            cum += hist[bin];
        }
        sel_hi = (unsigned)bin; rem_s = (unsigned)RANK_ - cum;
    }
    __syncthreads();
    const unsigned hi = sel_hi, rem = rem_s;
    for (int i = t; i < 65536; i += 1024) hist[i] = 0u;
    __syncthreads();
    for (int i = t; i < NN_; i += 1024) {
        unsigned bits = __float_as_uint(x[i]);
        if ((bits >> 16) == hi) atomicAdd(&hist[bits & 0xFFFFu], 1u);
    }
    __syncthreads();
    s = 0;
    for (int i = 0; i < 64; ++i) s += hist[t * 64 + i];
    psum[t] = s; __syncthreads();
    if (t == 0) {
        unsigned cum = 0; int chunk = 0;
        for (; chunk < 1024; ++chunk) {
            if (cum + psum[chunk] > rem) break;
            cum += psum[chunk];
        }
        int lo = chunk * 64;
        for (;; ++lo) {
            if (cum + hist[lo] > rem) break;
            cum += hist[lo];
        }
        thr[b] = __uint_as_float((hi << 16) | (unsigned)lo);
    }
}

// =====================================================================
// causal_adj = where(soft >= thr, soft, 0) / (sum + 1e-12)
// =====================================================================
__global__ __launch_bounds__(1024)
void mask_norm_k(const float* __restrict__ soft, const float* __restrict__ thr,
                 float* __restrict__ adj)
{
    __shared__ float red[1024];
    const int b = blockIdx.x, t = threadIdx.x;
    const float th = thr[b];
    const float* x = soft + (size_t)b * NN_;
    float* y = adj + (size_t)b * NN_;

    float sm = 0.0f;
    for (int i = t; i < NN_; i += 1024) { float v = x[i]; if (v >= th) sm += v; }
    red[t] = sm; __syncthreads();
    for (int s = 512; s > 0; s >>= 1) {
        if (t < s) red[t] += red[t + s];
        __syncthreads();
    }
    const float inv = 1.0f / (red[0] + 1e-12f);
    for (int i = t; i < NN_; i += 1024) {
        float v = x[i];
        y[i] = (v >= th) ? v * inv : 0.0f;
    }
}

// =====================================================================
// BatchNorm over (B,N): fold into per-channel scale/shift.
// =====================================================================
__global__ __launch_bounds__(128)
void bn_stats_k(const float* __restrict__ X, const float* __restrict__ gamma,
                const float* __restrict__ beta, float* __restrict__ scale,
                float* __restrict__ shift, int C)
{
    __shared__ float r1[128], r2[128];
    const int c = blockIdx.x, t = threadIdx.x;
    const int S = B_ * N_; // 1024
    float s1 = 0.0f, s2 = 0.0f;
    for (int i = t; i < S; i += 128) {
        float v = X[(size_t)i * C + c];
        s1 += v; s2 += v * v;
    }
    r1[t] = s1; r2[t] = s2; __syncthreads();
    for (int s = 64; s > 0; s >>= 1) {
        if (t < s) { r1[t] += r1[t + s]; r2[t] += r2[t + s]; }
        __syncthreads();
    }
    if (t == 0) {
        float m = r1[0] / (float)S;
        float var = r2[0] / (float)S - m * m;
        float sc = gamma[c] * rsqrtf(var + 1e-5f);
        scale[c] = sc;
        shift[c] = beta[c] - m * sc;
    }
}

__global__ __launch_bounds__(256)
void bn_apply_relu_k(float* __restrict__ X, const float* __restrict__ scale,
                     const float* __restrict__ shift, int C, int total)
{
    int idx = blockIdx.x * 256 + threadIdx.x;
    if (idx < total) {
        int c = idx % C;
        X[idx] = fmaxf(fmaf(X[idx], scale[c], shift[c]), 0.0f);
    }
}

// =====================================================================
// feat[b,:] = mean_n h5[b,n,:]; out[b,c] = feat . cls_w[c] + cls_b[c]
// =====================================================================
__global__ __launch_bounds__(128)
void pool_cls_k(const float* __restrict__ h5, const float* __restrict__ cw,
                const float* __restrict__ cb, float* __restrict__ out)
{
    __shared__ float feat[OUT_];
    const int b = blockIdx.x, t = threadIdx.x;
    float s = 0.0f;
    for (int n = 0; n < N_; ++n)
        s += h5[((size_t)b * N_ + n) * OUT_ + t];
    feat[t] = s * (1.0f / (float)N_);
    __syncthreads();
    if (t < NCLS_) {
        float acc = cb[t];
        for (int f = 0; f < OUT_; ++f) acc = fmaf(feat[f], cw[t * OUT_ + f], acc);
        out[b * NCLS_ + t] = acc;
    }
}

// =====================================================================
// Host orchestration
// =====================================================================
extern "C" void kernel_launch(void* const* d_in, const int* in_sizes, int n_in,
                              void* d_out, int out_size, void* d_ws, size_t ws_size,
                              hipStream_t stream)
{
    (void)in_sizes; (void)n_in; (void)out_size; (void)ws_size;
    const float* x    = (const float*)d_in[0];
    const float* Wg   = (const float*)d_in[1];
    const float* attn = (const float*)d_in[2];
    const float* W1   = (const float*)d_in[3];
    const float* b1   = (const float*)d_in[4];
    const float* w2   = (const float*)d_in[5];
    const float* b2   = (const float*)d_in[6];
    const float* gc1w = (const float*)d_in[7];
    const float* gc1b = (const float*)d_in[8];
    const float* bn1g = (const float*)d_in[9];
    const float* bn1b = (const float*)d_in[10];
    const float* gc2w = (const float*)d_in[11];
    const float* gc2b = (const float*)d_in[12];
    const float* bn2g = (const float*)d_in[13];
    const float* bn2b = (const float*)d_in[14];
    const float* clsw = (const float*)d_in[15];
    const float* clsb = (const float*)d_in[16];
    float* out = (float*)d_out;

    // ---- workspace layout (floats) ----
    float* W = (float*)d_ws;
    size_t off = 0;
    float* hp     = W + off; off += (size_t)B_ * N_ * FDIM_;
    float* node   = W + off; off += (size_t)B_ * N_ * FDIM_;
    float* s_i    = W + off; off += (size_t)B_ * HEADS_ * N_;
    float* s_j    = W + off; off += (size_t)B_ * HEADS_ * N_;
    float* alpha  = W + off; off += (size_t)B_ * HEADS_ * NN_;
    float* pi     = W + off; off += (size_t)B_ * N_ * AH_;
    float* pj     = W + off; off += (size_t)B_ * N_ * AH_;
    float* pjT    = W + off; off += (size_t)B_ * AH_ * N_;
    float* es     = W + off; off += (size_t)B_ * NN_;
    float* soft   = W + off; off += (size_t)B_ * NN_;
    float* adjm   = W + off; off += (size_t)B_ * NN_;
    float* h1     = W + off; off += (size_t)B_ * N_ * FDIM_;
    float* h2     = W + off; off += (size_t)B_ * N_ * HID_;
    float* h4     = W + off; off += (size_t)B_ * N_ * HID_;
    float* h5     = W + off; off += (size_t)B_ * N_ * OUT_;
    float* thr    = W + off; off += 64;
    float* scale1 = W + off; off += HID_;
    float* shift1 = W + off; off += HID_;
    float* scale2 = W + off; off += OUT_;
    float* shift2 = W + off; off += OUT_;

    const dim3 gb(256);

    // 1. hp = x @ Wg^T   (NT)  M=1024 N=1024 K=1024
    gemm_wmma_bf16<1, false><<<dim3(FDIM_ / 64, (B_ * N_) / 128, 1), gb, 0, stream>>>(
        x, Wg, hp, nullptr, B_ * N_, FDIM_, FDIM_, FDIM_, FDIM_, FDIM_,
        1, 0, 0, 0, 0, 0, 0);

    // 2. attention scores
    attn_scores_k<<<dim3(B_ * HEADS_ * N_), dim3(256), 0, stream>>>(hp, attn, s_i, s_j);

    // 3. alpha = softmax(leakyrelu(s_i + s_j))
    attn_softmax_k<<<dim3(B_ * HEADS_ * N_), dim3(512), 0, stream>>>(s_i, s_j, alpha);

    // 4. node_feats = alpha @ hp_head  (NN), per (b,h): M=512 N=256 K=512
    gemm_wmma_bf16<0, false><<<dim3(DH_ / 64, N_ / 128, B_ * HEADS_), gb, 0, stream>>>(
        alpha, hp, node, nullptr, N_, DH_, N_, N_, FDIM_, FDIM_,
        HEADS_,
        (long long)HEADS_ * NN_, (long long)NN_,
        (long long)N_ * FDIM_, (long long)DH_,
        (long long)N_ * FDIM_, (long long)DH_);

    // 5. pi = node @ W1i^T, pj = node @ W1j^T  (NT) M=1024 N=256 K=1024
    gemm_wmma_bf16<1, false><<<dim3(AH_ / 64, (B_ * N_) / 128, 1), gb, 0, stream>>>(
        node, W1, pi, nullptr, B_ * N_, AH_, FDIM_, FDIM_, 2 * FDIM_, AH_,
        1, 0, 0, 0, 0, 0, 0);
    gemm_wmma_bf16<1, false><<<dim3(AH_ / 64, (B_ * N_) / 128, 1), gb, 0, stream>>>(
        node, W1 + FDIM_, pj, nullptr, B_ * N_, AH_, FDIM_, FDIM_, 2 * FDIM_, AH_,
        1, 0, 0, 0, 0, 0, 0);

    // 6. pjT
    transpose_pj_k<<<dim3(N_ / 32, AH_ / 32, B_), dim3(32, 8), 0, stream>>>(pj, pjT);

    // 7. edge scores (fused MLP over AH)
    edge_scores_k<<<dim3(N_, B_), dim3(256), 0, stream>>>(pi, pjT, b1, w2, b2, es);

    // 8. flat softmax with temperature
    flat_softmax_k<<<dim3(B_), dim3(1024), 0, stream>>>(es, soft);

    // 9. exact threshold via 64K-bin LDS histogram (256KB dynamic LDS)
    select_thr_k<<<dim3(B_), dim3(1024), 65536 * sizeof(unsigned), stream>>>(soft, thr);

    // 10. mask + renormalize
    mask_norm_k<<<dim3(B_), dim3(1024), 0, stream>>>(soft, thr, adjm);

    // 11. h1 = adj @ node  (NN), per b: M=512 N=1024 K=512
    gemm_wmma_bf16<0, false><<<dim3(FDIM_ / 64, N_ / 128, B_), gb, 0, stream>>>(
        adjm, node, h1, nullptr, N_, FDIM_, N_, N_, FDIM_, FDIM_,
        1, (long long)NN_, 0, (long long)N_ * FDIM_, 0, (long long)N_ * FDIM_, 0);

    // 12. h2 = h1 @ gc1_w^T + gc1_b  (NT) M=1024 N=512 K=1024
    gemm_wmma_bf16<1, true><<<dim3(HID_ / 64, (B_ * N_) / 128, 1), gb, 0, stream>>>(
        h1, gc1w, h2, gc1b, B_ * N_, HID_, FDIM_, FDIM_, FDIM_, HID_,
        1, 0, 0, 0, 0, 0, 0);

    // 13. BN1 + relu (in place)
    bn_stats_k<<<dim3(HID_), dim3(128), 0, stream>>>(h2, bn1g, bn1b, scale1, shift1, HID_);
    bn_apply_relu_k<<<dim3((B_ * N_ * HID_) / 256), dim3(256), 0, stream>>>(
        h2, scale1, shift1, HID_, B_ * N_ * HID_);

    // 14. h4 = adj @ h2  (NN), per b: M=512 N=512 K=512
    gemm_wmma_bf16<0, false><<<dim3(HID_ / 64, N_ / 128, B_), gb, 0, stream>>>(
        adjm, h2, h4, nullptr, N_, HID_, N_, N_, HID_, HID_,
        1, (long long)NN_, 0, (long long)N_ * HID_, 0, (long long)N_ * HID_, 0);

    // 15. h5 = h4 @ gc2_w^T + gc2_b  (NT) M=1024 N=128 K=512
    gemm_wmma_bf16<1, true><<<dim3(OUT_ / 64, (B_ * N_) / 128, 1), gb, 0, stream>>>(
        h4, gc2w, h5, gc2b, B_ * N_, OUT_, HID_, HID_, HID_, OUT_,
        1, 0, 0, 0, 0, 0, 0);

    // 16. BN2 + relu (in place)
    bn_stats_k<<<dim3(OUT_), dim3(128), 0, stream>>>(h5, bn2g, bn2b, scale2, shift2, OUT_);
    bn_apply_relu_k<<<dim3((B_ * N_ * OUT_) / 256), dim3(256), 0, stream>>>(
        h5, scale2, shift2, OUT_, B_ * N_ * OUT_);

    // 17. mean pool + classifier
    pool_cls_k<<<dim3(B_), dim3(OUT_), 0, stream>>>(h5, clsw, clsb, out);
}
